// CondGraphConv_75952201663111
// MI455X (gfx1250) — matrix operand order repaired and checked by reading, not compiled, not run on hardware
//
#include <hip/hip_runtime.h>

typedef __attribute__((ext_vector_type(16))) _Float16 v16h;
typedef __attribute__((ext_vector_type(8)))  float    v8f;
typedef _Float16 f16;

#define N_NODES   6400
#define E_EDGES   313600
#define B_BATCH   128
#define NODE_DIM  2048
#define COND_DIM  1024
#define EDGE_DIM  128
#define S_IN      8
#define S_OUT     30
#define HID       286           // 2*EDGE_DIM + S_OUT
#define KC_NODE   64            // 2048 / 32
#define KC_EDGE   9             // 4 chunks hi, 4 chunks hj, 1 chunk s (padded to 32)
#define LN_EPS    1e-5f

// halves per B fragment-group of 8 chunks: 8 chunks * 8 tiles * 32 lanes * 16
#define GRP_HALVES (8 * 8 * 32 * 16)       // 32768 halves = 65536 bytes
#define GRP_UINT4  (GRP_HALVES / 8)        // 4096 16-byte vectors

// ---------------------------------------------------------------------------
// Fragment layout conventions (wave32, v_wmma_f32_16x16x32_f16):
//   A (16x32, f16):  lane l -> row m = l&15 ; khalf = l>>4
//                    halves 0..7  = K(chunk) = khalf*8 + 0..7
//                    halves 8..15 = K(chunk) = 16 + khalf*8 + 0..7
//                    => two contiguous 8-half (16B) runs at +0 and +16 halves.
//   B (32x16, f16):  lane l -> col n = l&15 ; halves 0..15 = same K order as A (khalf*8+0..7, 16+khalf*8+0..7)
//                    (pre-swizzled contiguous per lane: one v16h load)
//   C/D (16x16,f32): VGPR r: lanes 0-15 -> (M=r,   N=lane)
//                            lanes 16-31-> (M=r+8, N=lane-16)
// ---------------------------------------------------------------------------

// --- prep: swizzle W [K x 128] f32 row-major into fragment-major f16 --------
// frag[((c*8 + t)*32 + lane)*16 + k] = W[32c + khalf*8 + k + (k>=8 ? 8 : 0)][16t + (lane&15)]
__global__ void prep_wfrag(const float* __restrict__ W, f16* __restrict__ frag,
                           int kchunks, int krows) {
    int id = blockIdx.x * blockDim.x + threadIdx.x;   // = (c*8+t)*32 + lane
    if (id >= kchunks * 8 * 32) return;
    int lane = id & 31;
    int t    = (id >> 5) & 7;
    int c    = id >> 8;
    int n     = t * 16 + (lane & 15);
    int kbase = c * 32 + ((lane >> 4) << 3);
    f16* dst = frag + (size_t)id * 16;
#pragma unroll
    for (int k = 0; k < 16; ++k) {
        int kg = kbase + k + (k >= 8 ? 8 : 0);
        dst[k] = (kg < krows) ? (f16)W[(size_t)kg * EDGE_DIM + n] : (f16)0.0f;
    }
}

// --- prep: FiLM table GB[b][0..127]=gamma+1, GB[b][128..255]=beta -----------
__global__ void film_kernel(const float* __restrict__ cond,
                            const float* __restrict__ Wc,
                            const float* __restrict__ bc,
                            float* __restrict__ GB) {
    int id = blockIdx.x * blockDim.x + threadIdx.x;   // b*256 + col
    if (id >= B_BATCH * 2 * EDGE_DIM) return;
    int col = id & 255;
    int b   = id >> 8;
    const float* cf = cond + (size_t)b * COND_DIM;
    float acc = 0.0f;
    for (int k = 0; k < COND_DIM; ++k)
        acc = fmaf(cf[k], Wc[(size_t)k * (2 * EDGE_DIM) + col], acc);
    acc += bc[col];
    if (col < EDGE_DIM) acc += 1.0f;                  // gamma + 1
    GB[id] = acc;
}

// --- prep: s = relu(spatial @ Ws + bs), padded to 32 cols, f16 --------------
__global__ void spatial_kernel(const float* __restrict__ sp,
                               const float* __restrict__ Ws,
                               const float* __restrict__ bs,
                               f16* __restrict__ s16) {
    int e = blockIdx.x * blockDim.x + threadIdx.x;
    if (e >= E_EDGES) return;
    const float4* p = (const float4*)(sp + (size_t)e * S_IN);
    float4 a = p[0], b = p[1];
    float si[8] = {a.x, a.y, a.z, a.w, b.x, b.y, b.z, b.w};
    union { f16 h[32]; uint4 q[4]; } u;
#pragma unroll
    for (int j = 0; j < S_OUT; ++j) {
        float acc = bs[j];
#pragma unroll
        for (int k = 0; k < S_IN; ++k)
            acc = fmaf(si[k], Ws[k * S_OUT + j], acc);
        u.h[j] = (f16)fmaxf(acc, 0.0f);
    }
    u.h[30] = (f16)0.0f;
    u.h[31] = (f16)0.0f;
    uint4* dst = (uint4*)(s16 + (size_t)e * 32);
#pragma unroll
    for (int i = 0; i < 4; ++i) dst[i] = u.q[i];
}

// --- A-fragment loader (f16 source): two aligned 16B runs at +0/+16 halves --
__device__ __forceinline__ v16h load_a_frag(const f16* p) {
    union { uint4 q; f16 h[8]; } u0, u1;
    u0.q = *(const uint4*)(p);
    u1.q = *(const uint4*)(p + 16);
    v16h A;
#pragma unroll
    for (int k = 0; k < 8; ++k) { A[k] = u0.h[k]; A[k + 8] = u1.h[k]; }
    return A;
}

// --- A-fragment loader (f32 source): 4x float4 + convert --------------------
__device__ __forceinline__ v16h load_a_f32(const float* ap) {
    float4 a0 = *(const float4*)(ap);
    float4 a1 = *(const float4*)(ap + 4);
    float4 a2 = *(const float4*)(ap + 16);
    float4 a3 = *(const float4*)(ap + 20);
    v16h A;
    A[0]=(f16)a0.x; A[1]=(f16)a0.y; A[2] =(f16)a0.z; A[3] =(f16)a0.w;
    A[4]=(f16)a1.x; A[5]=(f16)a1.y; A[6] =(f16)a1.z; A[7] =(f16)a1.w;
    A[8]=(f16)a2.x; A[9]=(f16)a2.y; A[10]=(f16)a2.z; A[11]=(f16)a2.w;
    A[12]=(f16)a3.x;A[13]=(f16)a3.y;A[14]=(f16)a3.z; A[15]=(f16)a3.w;
    return A;
}

// --- node projection: h16 = relu(X @ Wn + bn), f16 output -------------------
// B fragments staged through LDS in 8 groups of 8 K-chunks (64KB per group);
// relaxed waves/EU so the allocator keeps a full chunk of B + prefetched A
// fragments live, enabling clause'd ds loads and partial dscnt waits.
__global__ void __launch_bounds__(256)
__attribute__((amdgpu_waves_per_eu(4)))
node_proj(const float* __restrict__ X, const f16* __restrict__ WnF,
          const float* __restrict__ bn, f16* __restrict__ h16) {
    extern __shared__ f16 lds[];
    int lane = threadIdx.x & 31;
    int wave = threadIdx.x >> 5;
    int gw   = blockIdx.x * 8 + wave;
    int r0   = gw * 16;
    int ml   = lane & 15;
    int kh   = lane >> 4;

    const v16h* BS = (const v16h*)lds;
    v8f acc[8] = {};

    const float* arow = X + (size_t)(r0 + ml) * NODE_DIM + kh * 8;
    for (int g = 0; g < KC_NODE / 8; ++g) {
        __syncthreads();                       // previous group fully consumed
        {
            const uint4* src = (const uint4*)(WnF + (size_t)g * GRP_HALVES);
            uint4* dst = (uint4*)lds;
            for (int i = threadIdx.x; i < GRP_UINT4; i += 256)
                dst[i] = src[i];
        }
        __syncthreads();

        v16h A = load_a_f32(arow + (g * 8) * 32);   // chunk 0 of this group
#pragma unroll
        for (int c8 = 0; c8 < 8; ++c8) {
            v16h B[8];
#pragma unroll
            for (int t = 0; t < 8; ++t)
                B[t] = BS[(c8 * 8 + t) * 32 + lane];
            v16h An = A;
            if (c8 < 7)                        // prefetch next chunk's A
                An = load_a_f32(arow + (g * 8 + c8 + 1) * 32);
#pragma unroll
            for (int t = 0; t < 8; ++t)
                acc[t] = __builtin_amdgcn_wmma_f32_16x16x32_f16(
                    false, A, false, B[t], (short)0, acc[t], false, false);
            A = An;
        }
    }
#pragma unroll
    for (int r = 0; r < 8; ++r) {
        int m = r + kh * 8;
        size_t rowoff = (size_t)(r0 + m) * EDGE_DIM;
#pragma unroll
        for (int t = 0; t < 8; ++t) {
            int n = t * 16 + ml;
            float v = acc[t][r] + bn[n];
            h16[rowoff + n] = (f16)fmaxf(v, 0.0f);
        }
    }
}

// --- main edge kernel: gather + GEMM (WMMA) + layernorm + FiLM + relu -------
// Wl fragments for K-chunks 0..7 staged once per block into LDS (64KB);
// per chunk: batch-load all 8 B fragments + prefetch next chunk's A, then
// run the 8-WMMA chain (partial waits, loads overlapped).
__global__ void __launch_bounds__(256)
__attribute__((amdgpu_waves_per_eu(4)))
edge_kernel(const f16* __restrict__ h16, const f16* __restrict__ s16,
            const f16* __restrict__ WlF, const float* __restrict__ GB,
            const int* __restrict__ ni, const int* __restrict__ nj,
            const int* __restrict__ bid, float* __restrict__ out) {
    extern __shared__ f16 lds[];
    int lane = threadIdx.x & 31;
    int wave = threadIdx.x >> 5;
    int gw   = blockIdx.x * 8 + wave;
    int e0   = gw * 16;
    int ml   = lane & 15;
    int kh   = lane >> 4;

    // stage B fragments for chunks 0..7 (hi & hj weights) into LDS
    {
        const uint4* src = (const uint4*)WlF;
        uint4* dst = (uint4*)lds;
        for (int i = threadIdx.x; i < GRP_UINT4; i += 256)
            dst[i] = src[i];
    }

    int e  = e0 + ml;
    int ii = ni[e];
    int jj = nj[e];
    int eb = bid[jj];                      // FiLM batch id for this edge row

    const f16* hi = h16 + (size_t)ii * EDGE_DIM + kh * 8;
    const f16* hj = h16 + (size_t)jj * EDGE_DIM + kh * 8;
    const f16* sp = s16 + (size_t)e  * 32       + kh * 8;
    const v16h* BS  = (const v16h*)lds;
    const v16h* WlV = (const v16h*)WlF;

    __syncthreads();

    v8f acc[8] = {};

    v16h A = load_a_frag(hi);              // chunk 0
#pragma unroll
    for (int cc = 0; cc < KC_EDGE; ++cc) { // 0..3: hi, 4..7: hj, 8: spatial
        v16h B[8];
#pragma unroll
        for (int t = 0; t < 8; ++t)
            B[t] = (cc < 8) ? BS[(cc * 8 + t) * 32 + lane]
                            : WlV[(8 * 8 + t) * 32 + lane];
        v16h An = A;
        if (cc < 8) {                      // prefetch next chunk's A
            int cn = cc + 1;
            const f16* ap = (cn < 4) ? (hi + cn * 32)
                          : (cn < 8) ? (hj + (cn - 4) * 32)
                                     : sp;
            An = load_a_frag(ap);
        }
#pragma unroll
        for (int t = 0; t < 8; ++t)
            acc[t] = __builtin_amdgcn_wmma_f32_16x16x32_f16(
                false, A, false, B[t], (short)0, acc[t], false, false);
        A = An;
    }

    // Epilogue: per-row layernorm (rows of 128 = 8 tiles x 16 lanes of a half)
    const float inv = 1.0f / 128.0f;
#pragma unroll
    for (int r = 0; r < 8; ++r) {
        float s = 0.0f, q = 0.0f;
#pragma unroll
        for (int t = 0; t < 8; ++t) { float x = acc[t][r]; s += x; q += x * x; }
#pragma unroll
        for (int d = 1; d < 16; d <<= 1) {     // butterfly within 16-lane half
            s += __shfl_xor(s, d, 32);
            q += __shfl_xor(q, d, 32);
        }
        float mean = s * inv;
        float var  = q * inv - mean * mean;
        float rstd = rsqrtf(var + LN_EPS);
        int m = r + kh * 8;                    // actual edge row of this slot
        int ebr = __shfl(eb, m, 32);           // eb for row m lives at lane m
        const float* g = GB + (size_t)ebr * 256;
        size_t rowoff = (size_t)(e0 + m) * EDGE_DIM;
#pragma unroll
        for (int t = 0; t < 8; ++t) {
            int n = t * 16 + ml;
            float x = (acc[t][r] - mean) * rstd;
            float y = fmaf(x, g[n], g[128 + n]);   // x*(gamma+1) + beta
            out[rowoff + n] = fmaxf(y, 0.0f);
        }
    }
}

// ---------------------------------------------------------------------------
extern "C" void kernel_launch(void* const* d_in, const int* in_sizes, int n_in,
                              void* d_out, int out_size, void* d_ws, size_t ws_size,
                              hipStream_t stream) {
    const float* node_feats    = (const float*)d_in[0];
    const float* cond_feats    = (const float*)d_in[1];
    const float* spatial_feats = (const float*)d_in[2];
    const float* Wn            = (const float*)d_in[3];
    const float* bn            = (const float*)d_in[4];
    const float* Ws            = (const float*)d_in[5];
    const float* bs            = (const float*)d_in[6];
    const float* Wc            = (const float*)d_in[7];
    const float* bc            = (const float*)d_in[8];
    const float* Wl            = (const float*)d_in[9];
    const int*   node_i_ids    = (const int*)d_in[10];
    const int*   node_j_ids    = (const int*)d_in[11];
    const int*   batch_ids     = (const int*)d_in[12];
    float*       out           = (float*)d_out;

    // workspace partition (256B aligned)
    char* ws = (char*)d_ws;
    size_t off = 0;
    auto take = [&](size_t bytes) {
        char* p = ws + off;
        off = (off + bytes + 255) & ~(size_t)255;
        return p;
    };
    f16*   h16  = (f16*)  take((size_t)N_NODES * EDGE_DIM * sizeof(f16));
    f16*   s16  = (f16*)  take((size_t)E_EDGES * 32 * sizeof(f16));
    f16*   WnF  = (f16*)  take((size_t)KC_NODE * 8 * 32 * 16 * sizeof(f16));
    f16*   WlF  = (f16*)  take((size_t)KC_EDGE * 8 * 32 * 16 * sizeof(f16));
    float* GB   = (float*)take((size_t)B_BATCH * 2 * EDGE_DIM * sizeof(float));
    (void)ws_size; (void)in_sizes; (void)n_in; (void)out_size;

    // prep: weight fragment swizzles
    {
        int n = KC_NODE * 8 * 32;
        prep_wfrag<<<(n + 255) / 256, 256, 0, stream>>>(Wn, WnF, KC_NODE, NODE_DIM);
    }
    {
        int n = KC_EDGE * 8 * 32;
        prep_wfrag<<<(n + 255) / 256, 256, 0, stream>>>(Wl, WlF, KC_EDGE, HID);
    }
    // prep: FiLM gamma+1 / beta table
    {
        int n = B_BATCH * 2 * EDGE_DIM;
        film_kernel<<<(n + 255) / 256, 256, 0, stream>>>(cond_feats, Wc, bc, GB);
    }
    // prep: spatial features, f16 padded to 32
    spatial_kernel<<<(E_EDGES + 255) / 256, 256, 0, stream>>>(spatial_feats, Ws, bs, s16);

    // node projection: 6400 rows / (16 rows * 8 waves) = 50 blocks, 64KB LDS
    node_proj<<<N_NODES / (16 * 8), 256, GRP_UINT4 * sizeof(uint4), stream>>>(
        node_feats, WnF, bn, h16);

    // edge GEMM + LN + FiLM: 2450 blocks, 64KB LDS for staged Wl fragments
    edge_kernel<<<E_EDGES / (16 * 8), 256, GRP_UINT4 * sizeof(uint4), stream>>>(
        h16, s16, WlF, GB, node_i_ids, node_j_ids, batch_ids, out);
}
